// YangMillsNet_80590766343028
// MI455X (gfx1250) — compile-verified, matrix operand order checked by reading, not run on hardware
//
#include <hip/hip_runtime.h>

// ---------------------------------------------------------------------------
// YangMillsNet fused forward for MI455X (gfx1250), wave32 + WMMA f32_16x16x32_f16
// ---------------------------------------------------------------------------

typedef __attribute__((ext_vector_type(16))) _Float16 v16h;
typedef __attribute__((ext_vector_type(8)))  float    v8f;
typedef __attribute__((ext_vector_type(4)))  float    v4f;   // native vector for NT stores

#define BDIM   256
#define RS     1028      // padded f32 row stride (floats) for the 16x1024 H tile
#define XS_RS  1032      // padded f16 row stride (halves) for the X tile
#define FS_RS  168       // padded f16 row stride for Fourier features (K padded 160)
#define EPSN   1e-6f

// K index inside a 16-bit A/B fragment for element e=(2v+p), lane-half `half`
// (ISA 7.12.2: VGPR0..3 -> K = 8*half + 2v+p ; VGPR4..7 -> K = 16 + 8*half + 2(v-4)+p)
__device__ __host__ __forceinline__ int kof(int v, int p, int half) {
  return ((v < 4) ? (2 * v) : (16 + 2 * (v - 4))) + 8 * half + p;
}

// Cl(3,0) geometric product, basis [1,e1,e2,e3,e12,e13,e23,e123] (bitmask order
// 0,1,2,4,3,5,6,7).  Fully unrolled -> Cayley table constant-folds.
__device__ __forceinline__ void gp8(const float a[8], const float b[8], float r[8]) {
  const int bl[8]  = {0, 1, 2, 4, 3, 5, 6, 7};
  const int inv[8] = {0, 1, 2, 4, 3, 5, 6, 7};  // inv[mask] = index
#pragma unroll
  for (int k = 0; k < 8; k++) r[k] = 0.f;
#pragma unroll
  for (int i = 0; i < 8; i++) {
#pragma unroll
    for (int j = 0; j < 8; j++) {
      const int A = bl[i], Bm = bl[j];
      const int s = __popc((A >> 1) & Bm) + __popc((A >> 2) & Bm);
      const float sg = (s & 1) ? -1.f : 1.f;
      r[inv[A ^ Bm]] += sg * a[i] * b[j];
    }
  }
}

// In-place grade-wise RMS norm of the 16x(128x8) f32 LDS tile.
__device__ __forceinline__ void grade_norm_ip(float* Hs, float* red,
                                              const float* __restrict__ gamma,
                                              int tid) {
  if (tid < 64) red[tid] = 0.f;
  __syncthreads();
  const int row = tid >> 4, c0 = (tid & 15) << 3;
  float s0 = 0, s1 = 0, s2 = 0, s3 = 0;
#pragma unroll
  for (int j = 0; j < 8; j++) {
    const float* p = &Hs[row * RS + ((c0 + j) << 3)];
    s0 += p[0] * p[0];
    s1 += p[1] * p[1] + p[2] * p[2] + p[3] * p[3];
    s2 += p[4] * p[4] + p[5] * p[5] + p[6] * p[6];
    s3 += p[7] * p[7];
  }
  atomicAdd(&red[(row << 2) + 0], s0);
  atomicAdd(&red[(row << 2) + 1], s1);
  atomicAdd(&red[(row << 2) + 2], s2);
  atomicAdd(&red[(row << 2) + 3], s3);
  __syncthreads();
  float inv[4];
#pragma unroll
  for (int g = 0; g < 4; g++)
    inv[g] = 1.f / sqrtf(red[(row << 2) + g] * 0.0078125f + EPSN);
#pragma unroll
  for (int j = 0; j < 8; j++) {
    const int c = c0 + j;
    float* p = &Hs[row * RS + (c << 3)];
    const float g0 = gamma[(c << 2) + 0] * inv[0];
    const float g1 = gamma[(c << 2) + 1] * inv[1];
    const float g2 = gamma[(c << 2) + 2] * inv[2];
    const float g3 = gamma[(c << 2) + 3] * inv[3];
    p[0] *= g0;
    p[1] *= g1; p[2] *= g1; p[3] *= g1;
    p[4] *= g2; p[5] *= g2; p[6] *= g2;
    p[7] *= g3;
  }
  __syncthreads();
}

// ---------------------------------------------------------------------------
// Weight pre-pack: lin_W [8,128,128,8] f32 -> WMMA-B-fragment-ordered f16.
// Layout: ((((l*8+k)*8+nt)*4+kk)*32+lane)*16+e  (one coalesced 32B/lane blob).
// ---------------------------------------------------------------------------
__global__ void prep_linw(const float* __restrict__ lin_W, _Float16* __restrict__ w16) {
  const int idx = blockIdx.x * blockDim.x + threadIdx.x;
  if (idx >= (1 << 20)) return;
  const int e    = idx & 15;
  const int lane = (idx >> 4) & 31;
  const int kk   = (idx >> 9) & 3;
  const int nt   = (idx >> 11) & 7;
  const int k    = (idx >> 14) & 7;
  const int l    = idx >> 17;
  const int o = nt * 16 + (lane & 15);
  const int K = kk * 32 + kof(e >> 1, e & 1, lane >> 4);
  w16[idx] = (_Float16)lin_W[((l * 128 + o) * 128 + K) * 8 + k];
}

// lift_W [132,1024] f32 -> B fragments with K padded to 160 (zeros for K>=132).
// Layout: ((nt*5+kk)*32+lane)*16+e, nt in 0..63.
__global__ void prep_liftw(const float* __restrict__ lift_W, _Float16* __restrict__ w16) {
  const int idx = blockIdx.x * blockDim.x + threadIdx.x;
  if (idx >= 64 * 5 * 32 * 16) return;
  const int e    = idx & 15;
  const int lane = (idx >> 4) & 31;
  const int t3   = idx >> 9;
  const int kk   = t3 % 5;
  const int nt   = t3 / 5;
  const int o = nt * 16 + (lane & 15);
  const int K = kk * 32 + kof(e >> 1, e & 1, lane >> 4);
  w16[idx] = (K < 132) ? (_Float16)lift_W[K * 1024 + o] : (_Float16)0.f;
}

// ---------------------------------------------------------------------------
// Fused network: one block = 16 rows of B, carried through all layers in LDS.
// wave w (0..7) owns blade w for all clifford_linear matmuls.
// ---------------------------------------------------------------------------
__global__ __launch_bounds__(BDIM)
void ym_main(const float* __restrict__ coords, const float* __restrict__ freq,
             const float* __restrict__ lift_b, const float* __restrict__ in_gamma,
             const float* __restrict__ norm_gamma, const float* __restrict__ theta,
             const float* __restrict__ alpha, const float* __restrict__ lin_b,
             const float* __restrict__ out_gamma, const float* __restrict__ projW,
             const float* __restrict__ projb,
             const _Float16* __restrict__ wlin, const _Float16* __restrict__ wlift,
             float* __restrict__ out) {
  __shared__ float    Hs[16 * RS];        // current h (f32, residual carrier)
  __shared__ _Float16 Xs[16 * XS_RS];     // transformed activations (f16, WMMA A)
  __shared__ _Float16 Fs[16 * FS_RS];     // Fourier features, K padded to 160
  __shared__ float    red[64];            // per-(row,grade) reduction scratch
  __shared__ float    rot[128 * 4];       // per-channel rotor coeffs {ca, r4, r5, r6}

  const int tid  = threadIdx.x;
  const int lane = tid & 31;
  const int wv   = tid >> 5;              // wave id == blade id
  const int half = lane >> 4;
  const int nn   = lane & 15;
  const int b0   = blockIdx.x * 16;

  // ---- Fourier features: feats = [coords, sin(coords@fb), cos(coords@fb)] ----
  for (int idx = tid; idx < 16 * FS_RS; idx += BDIM) {
    const int m = idx / FS_RS, q = idx - m * FS_RS;
    float v = 0.f;
    if (q < 4) {
      v = coords[(b0 + m) * 4 + q];
    } else if (q < 132) {
      const int f = (q - 4) & 63;
      float d = 0.f;
#pragma unroll
      for (int i = 0; i < 4; i++) d += coords[(b0 + m) * 4 + i] * freq[i * 64 + f];
      v = (q < 68) ? sinf(d) : cosf(d);
    }
    Fs[idx] = (_Float16)v;
  }
  __syncthreads();

  // ---- Lift: H[m][o] = feats @ lift_W + lift_b (o = c*8+k, 64 N-tiles) ----
  {
    v16h A[5];
#pragma unroll
    for (int kk = 0; kk < 5; kk++)
#pragma unroll
      for (int e = 0; e < 16; e++)
        A[kk][e] = Fs[nn * FS_RS + kk * 32 + kof(e >> 1, e & 1, half)];
    for (int t = 0; t < 8; t++) {
      const int nt = wv * 8 + t;
      v8f acc = {0.f, 0.f, 0.f, 0.f, 0.f, 0.f, 0.f, 0.f};
#pragma unroll
      for (int kk = 0; kk < 5; kk++) {
        v16h Bf = *(const v16h*)(wlift + ((nt * 5 + kk) << 9) + (lane << 4));
        acc = __builtin_amdgcn_wmma_f32_16x16x32_f16(false, A[kk], false, Bf,
                                                     (short)0, acc, false, false);
      }
      const int o = nt * 16 + nn;
#pragma unroll
      for (int r = 0; r < 8; r++)
        Hs[(r + 8 * half) * RS + o] = acc[r] + lift_b[o];
    }
  }
  __syncthreads();

  grade_norm_ip(Hs, red, in_gamma, tid);

  // ---- 8 residual Clifford layers ----
  for (int l = 0; l < 8; l++) {
    // rotor coefficients: channel-only -> compute once per layer (TRANS-pipe
    // work reduced 16x); overlapped with reduction-scratch zeroing.
    if (tid < 64) red[tid] = 0.f;
    if (tid < 128) {
      const float* tp = &theta[(l * 128 + tid) * 3];
      const float t0 = tp[0], t1 = tp[1], t2 = tp[2];
      const float ang = sqrtf(t0 * t0 + t1 * t1 + t2 * t2 + EPSN);
      const float sia = sinf(ang) / ang;
      rot[(tid << 2) + 0] = cosf(ang);
      rot[(tid << 2) + 1] = sia * t0;
      rot[(tid << 2) + 2] = sia * t1;
      rot[(tid << 2) + 3] = sia * t2;
    }
    __syncthreads();
    const int row = tid >> 4, c0 = (tid & 15) << 3;
    {
      float s0 = 0, s1 = 0, s2 = 0, s3 = 0;
#pragma unroll
      for (int j = 0; j < 8; j++) {
        const float* p = &Hs[row * RS + ((c0 + j) << 3)];
        s0 += p[0] * p[0];
        s1 += p[1] * p[1] + p[2] * p[2] + p[3] * p[3];
        s2 += p[4] * p[4] + p[5] * p[5] + p[6] * p[6];
        s3 += p[7] * p[7];
      }
      atomicAdd(&red[(row << 2) + 0], s0);
      atomicAdd(&red[(row << 2) + 1], s1);
      atomicAdd(&red[(row << 2) + 2], s2);
      atomicAdd(&red[(row << 2) + 3], s3);
    }
    __syncthreads();
    {
      float inv[4];
#pragma unroll
      for (int g = 0; g < 4; g++)
        inv[g] = 1.f / sqrtf(red[(row << 2) + g] * 0.0078125f + EPSN);
#pragma unroll
      for (int j = 0; j < 8; j++) {
        const int c = c0 + j;
        const float* p  = &Hs[row * RS + (c << 3)];
        const float* gm = &norm_gamma[(l * 128 + c) << 2];
        float x[8];
        x[0] = p[0] * gm[0] * inv[0];
        x[1] = p[1] * gm[1] * inv[1];
        x[2] = p[2] * gm[1] * inv[1];
        x[3] = p[3] * gm[1] * inv[1];
        x[4] = p[4] * gm[2] * inv[2];
        x[5] = p[5] * gm[2] * inv[2];
        x[6] = p[6] * gm[2] * inv[2];
        x[7] = p[7] * gm[3] * inv[3];
        // rotor sandwich y = R x ~R with precomputed per-channel R
        const float* rc = &rot[c << 2];
        const float R[8]  = {rc[0], 0.f, 0.f, 0.f, rc[1], rc[2], rc[3], 0.f};
        const float Rr[8] = {rc[0], 0.f, 0.f, 0.f, -rc[1], -rc[2], -rc[3], 0.f};
        float tm[8], y[8];
        gp8(R, x, tm);
        gp8(tm, Rr, y);
        // geometric GELU gate on multivector norm
        float n2 = EPSN;
#pragma unroll
        for (int k = 0; k < 8; k++) n2 += y[k] * y[k];
        const float gate =
            0.5f * (1.f + erff(alpha[l * 128 + c] * sqrtf(n2) * 0.70710678118654752f));
        _Float16* q = &Xs[row * XS_RS + (c << 3)];
#pragma unroll
        for (int k = 0; k < 8; k++) q[k] = (_Float16)(y[k] * gate);
      }
    }
    __syncthreads();

    // matmul: H += X @ W_l(blade wv) + lin_b   (wave wv owns blade wv)
    {
      v16h A[4];
#pragma unroll
      for (int kk = 0; kk < 4; kk++)
#pragma unroll
        for (int e = 0; e < 16; e++)
          A[kk][e] = Xs[nn * XS_RS + ((kk * 32 + kof(e >> 1, e & 1, half)) << 3) + wv];
      const _Float16* wb = wlin + ((size_t)(l * 8 + wv) << 14);
#pragma unroll
      for (int nt = 0; nt < 8; nt++) {
        v8f acc = {0.f, 0.f, 0.f, 0.f, 0.f, 0.f, 0.f, 0.f};
#pragma unroll
        for (int kk = 0; kk < 4; kk++) {
          v16h Bf = *(const v16h*)(wb + ((nt * 4 + kk) << 9) + (lane << 4));
          acc = __builtin_amdgcn_wmma_f32_16x16x32_f16(false, A[kk], false, Bf,
                                                       (short)0, acc, false, false);
        }
        const int o = nt * 16 + nn;  // output channel
        const float bia = lin_b[(l * 128 + o) * 8 + wv];
#pragma unroll
        for (int r = 0; r < 8; r++) {
          const int ad = (r + 8 * half) * RS + (o << 3) + wv;
          Hs[ad] = Hs[ad] + acc[r] + bia;
        }
      }
    }
    __syncthreads();

    // streamed intermediate store [L,B,C,8]: non-temporal so the 268 MB of
    // write-once data rinses through L2 without evicting the weight fragments.
    {
      float* dst = out + 262144 + (((size_t)l * 8192 + b0) << 10);
      for (int idx = tid; idx < 4096; idx += BDIM) {
        const int rr = idx >> 8, off = idx & 255;
        const v4f val = *(const v4f*)&Hs[rr * RS + (off << 2)];
        __builtin_nontemporal_store(val, (v4f*)(dst + (rr << 10) + (off << 2)));
      }
    }
    __syncthreads();
  }

  grade_norm_ip(Hs, red, out_gamma, tid);

  // ---- projection to A_mu [B,4,8], bivector-masked ----
  for (int idx = tid; idx < 512; idx += BDIM) {
    const int rr = idx >> 5, mu = (idx >> 3) & 3, k = idx & 7;
    float acc = projb[mu * 8 + k];
    for (int c = 0; c < 128; c++)
      acc += Hs[rr * RS + (c << 3) + k] * projW[(mu * 128 + c) * 8 + k];
    if (k < 4 || k > 6) acc = 0.f;  // keep grade-2 only
    out[((b0 + rr) << 5) + mu * 8 + k] = acc;
  }
}

// ---------------------------------------------------------------------------
extern "C" void kernel_launch(void* const* d_in, const int* in_sizes, int n_in,
                              void* d_out, int out_size, void* d_ws, size_t ws_size,
                              hipStream_t stream) {
  (void)in_sizes; (void)n_in; (void)out_size; (void)ws_size;
  const float* coords    = (const float*)d_in[0];
  const float* freq      = (const float*)d_in[1];
  const float* lift_W    = (const float*)d_in[2];
  const float* lift_b    = (const float*)d_in[3];
  const float* in_gamma  = (const float*)d_in[4];
  const float* n_gamma   = (const float*)d_in[5];
  const float* theta     = (const float*)d_in[6];
  const float* alpha     = (const float*)d_in[7];
  const float* lin_W     = (const float*)d_in[8];
  const float* lin_b     = (const float*)d_in[9];
  const float* out_gamma = (const float*)d_in[10];
  const float* proj_W    = (const float*)d_in[11];
  const float* proj_b    = (const float*)d_in[12];
  float* out = (float*)d_out;

  _Float16* wlin  = (_Float16*)d_ws;                                   // 2 MB
  _Float16* wlift = (_Float16*)((char*)d_ws + (2u << 20));             // 320 KB

  prep_linw<<<4096, BDIM, 0, stream>>>(lin_W, wlin);
  prep_liftw<<<640, BDIM, 0, stream>>>(lift_W, wlift);
  ym_main<<<512, BDIM, 0, stream>>>(coords, freq, lift_b, in_gamma, n_gamma, theta,
                                    alpha, lin_b, out_gamma, proj_W, proj_b,
                                    wlin, wlift, out);
}
